// StructureGuidedAttention_54314156425233
// MI455X (gfx1250) — matrix operand order repaired and checked by reading, not compiled, not run on hardware
//
#include <hip/hip_runtime.h>
#include <hip/hip_bf16.h>
#include <stdint.h>

#define D_MODEL 1024
#define N_HEADS 16
#define DK      64
#define BATCH   4
#define SEQ     1024
#define GEOMD   7

typedef __attribute__((ext_vector_type(16))) __bf16 v16bf;
typedef __attribute__((ext_vector_type(8)))  float  v8f;
typedef __attribute__((ext_vector_type(4)))  int    v4i;

union ABReg {
    v16bf   v;
    uint4   q[2];
    unsigned u[8];
};

__device__ __forceinline__ unsigned short f2bf(float f) {
    unsigned u = __builtin_bit_cast(unsigned, f);
    u += 0x7FFFu + ((u >> 16) & 1u);          // round-to-nearest-even
    return (unsigned short)(u >> 16);
}
__device__ __forceinline__ unsigned pack2(float a, float b) {
    return (unsigned)f2bf(a) | ((unsigned)f2bf(b) << 16);
}

// Async 16B global->LDS copy (per lane), tracked by ASYNCcnt.
__device__ __forceinline__ void async_copy16(const unsigned short* gsrc,
                                             unsigned short* ldst)
{
#if __has_builtin(__builtin_amdgcn_global_load_async_to_lds_b128)
    __builtin_amdgcn_global_load_async_to_lds_b128(
        (__attribute__((address_space(1))) v4i*)gsrc,
        (__attribute__((address_space(3))) v4i*)ldst,
        0, 0);
#else
    unsigned lofs =
        (unsigned)(size_t)((__attribute__((address_space(3))) unsigned short*)ldst);
    asm volatile("global_load_async_to_lds_b128 %0, %1, off"
                 :: "v"(lofs), "v"(gsrc) : "memory");
#endif
}
__device__ __forceinline__ void wait_async0() {
    asm volatile("s_wait_asynccnt 0x0" ::: "memory");
}

// ---------------------------------------------------------------------------
// Generic 64x128-per-block GEMM:  Out = A[4096,1024] @ W[1024,1024] + bias
// AMODE: 0 = A is f32 (convert on the fly), 1 = A is bf16
// OMODE: 0 = f32 row-major [4096,1024]
//        1 = bf16 head-split [B,H,S,DK]      (Q, K)
//        2 = bf16 head-split transposed [B,H,DK,S]  (V^T)
// 256 threads = 8 waves; wave w -> 16-row x 64-col sub-tile (4 WMMA C tiles).
// ---------------------------------------------------------------------------
#define OM_F32 0
#define OM_QK  1
#define OM_VT  2

template <int AMODE, int OMODE>
__global__ __launch_bounds__(256)
void gemm_proj(const void* __restrict__ Aptr, const float* __restrict__ W,
               const float* __restrict__ bias, void* __restrict__ Out)
{
    // W tile staged transposed: Wt[n(128)][k(32)] bf16, row stride 40 halves
    // (80 B) -> 16 lanes reading 16 different n hit 16 distinct banks.
    __shared__ unsigned short Wt[128 * 40];

    const int lane = threadIdx.x & 31;
    const int wave = threadIdx.x >> 5;
    const int hi   = lane >> 4;       // lane half (K-split selector)
    const int ln   = lane & 15;
    const int m0   = blockIdx.x * 64 + (wave >> 1) * 16;
    const int nblk = blockIdx.y * 128;
    const int n0   = nblk + (wave & 1) * 64;
    const int kOff = hi * 8;

    const float*          A32 = (const float*)Aptr;
    const unsigned short* A16 = (const unsigned short*)Aptr;

    v8f acc[4];
    {
        v8f z = {};
        acc[0] = z; acc[1] = z; acc[2] = z; acc[3] = z;
    }

    for (int kc = 0; kc < D_MODEL; kc += 32) {
        // ---- cooperative staging of W[kc:kc+32, nblk:nblk+128] (transposed)
        {
            const int rowT = threadIdx.x >> 3;   // 0..31  (k within tile)
            const int seg  = threadIdx.x & 7;    // 0..7   (16 columns each)
            const float* wrow = W + (size_t)(kc + rowT) * D_MODEL + nblk + seg * 16;
#pragma unroll
            for (int i = 0; i < 4; ++i) {
                float4 w4 = *(const float4*)(wrow + i * 4);
                const int nloc = seg * 16 + i * 4;
                Wt[(nloc + 0) * 40 + rowT] = f2bf(w4.x);
                Wt[(nloc + 1) * 40 + rowT] = f2bf(w4.y);
                Wt[(nloc + 2) * 40 + rowT] = f2bf(w4.z);
                Wt[(nloc + 3) * 40 + rowT] = f2bf(w4.w);
            }
        }
        __syncthreads();

        // ---- A fragment 16x32 (WMMA A layout, 2x b128 per lane when bf16)
        ABReg a;
        if (AMODE == 0) {
            const float* base = A32 + (size_t)(m0 + ln) * D_MODEL + kc + kOff;
            float4 f0 = *(const float4*)(base + 0);
            float4 f1 = *(const float4*)(base + 4);
            float4 f2 = *(const float4*)(base + 16);
            float4 f3 = *(const float4*)(base + 20);
            a.u[0] = pack2(f0.x, f0.y); a.u[1] = pack2(f0.z, f0.w);
            a.u[2] = pack2(f1.x, f1.y); a.u[3] = pack2(f1.z, f1.w);
            a.u[4] = pack2(f2.x, f2.y); a.u[5] = pack2(f2.z, f2.w);
            a.u[6] = pack2(f3.x, f3.y); a.u[7] = pack2(f3.z, f3.w);
        } else {
            const unsigned short* base =
                A16 + (size_t)(m0 + ln) * D_MODEL + kc + kOff;
            a.q[0] = *(const uint4*)(base);
            a.q[1] = *(const uint4*)(base + 16);
        }

        // ---- 4 B fragments from LDS + 4 WMMAs
#pragma unroll
        for (int ns = 0; ns < 4; ++ns) {
            ABReg bm;
            const unsigned short* bp =
                &Wt[((wave & 1) * 64 + ns * 16 + ln) * 40 + hi * 8];
            bm.q[0] = *(const uint4*)(bp);
            bm.q[1] = *(const uint4*)(bp + 16);
            acc[ns] = __builtin_amdgcn_wmma_f32_16x16x32_bf16(
                false, a.v, false, bm.v, (short)0, acc[ns], false, false);
        }
        __syncthreads();   // protect Wt before next staging round
    }

    // ---- epilogue: bias + layout-specific store
    const int row_lo = m0 + hi * 8;
#pragma unroll
    for (int ns = 0; ns < 4; ++ns) {
        const int col = n0 + ns * 16 + ln;
        const float bv = bias[col];
        if (OMODE == OM_F32) {
            float* o = (float*)Out;
#pragma unroll
            for (int r = 0; r < 8; ++r)
                o[(size_t)(row_lo + r) * D_MODEL + col] = acc[ns][r] + bv;
        } else if (OMODE == OM_QK) {
            unsigned short* o = (unsigned short*)Out;
            const int h = col >> 6, d = col & 63;
#pragma unroll
            for (int r = 0; r < 8; ++r) {
                const int row = row_lo + r;
                const int b = row >> 10, s = row & 1023;
                o[((size_t)(b * N_HEADS + h) * SEQ + s) * DK + d] =
                    f2bf(acc[ns][r] + bv);
            }
        } else { // OM_VT
            unsigned short* o = (unsigned short*)Out;
            const int h = col >> 6, d = col & 63;
#pragma unroll
            for (int r = 0; r < 8; ++r) {
                const int row = row_lo + r;
                const int b = row >> 10, s = row & 1023;
                o[((size_t)(b * N_HEADS + h) * DK + d) * SEQ + s] =
                    f2bf(acc[ns][r] + bv);
            }
        }
    }
}

// ---------------------------------------------------------------------------
// Fused flash attention with geometric bias + key-padding mask.
// Grid: (S/128, H, B); 256 threads = 8 waves; wave -> 16 q-rows.
// K/V tiles staged once per workgroup via async global->LDS copies.
// ---------------------------------------------------------------------------
__global__ __launch_bounds__(256)
void flash_attn_geom(const unsigned short* __restrict__ Q,
                     const unsigned short* __restrict__ K,
                     const unsigned short* __restrict__ Vt,
                     const float* __restrict__ geom,
                     const int*   __restrict__ mask,
                     const float* __restrict__ Wg,
                     const float* __restrict__ bg,
                     unsigned short* __restrict__ ctx)
{
    // K tile: 32 s-rows x 64 d (stride 72 halves -> conflict-free b128 reads)
    // V tile: 64 d-rows x 32 kk (stride 40 halves)
    __shared__ unsigned short Kl[32 * 72];
    __shared__ unsigned short Vl[64 * 40];
    __shared__ unsigned short Plds[8][16 * 32];   // per-wave P staging (1 KB)

    const int lane = threadIdx.x & 31;
    const int wave = threadIdx.x >> 5;
    const int hi   = lane >> 4;
    const int ln   = lane & 15;
    const int b    = blockIdx.z;
    const int h    = blockIdx.y;
    const int q0   = blockIdx.x * 128 + wave * 16;

    const size_t headOff = (size_t)(b * N_HEADS + h) * SEQ * DK;
    const unsigned short* Qh = Q  + headOff;
    const unsigned short* Kh = K  + headOff;
    const unsigned short* Vh = Vt + headOff;      // [d][s]

    float wg[GEOMD];
#pragma unroll
    for (int i = 0; i < GEOMD; ++i) wg[i] = Wg[i * N_HEADS + h];
    const float bgh = bg[h];
    const float inv_scale = 0.125f;               // 1/sqrt(64)

    // Q fragments for dk = 0..31 and 32..63
    ABReg qa[2];
    {
        const unsigned short* qb = Qh + (size_t)(q0 + ln) * DK + hi * 8;
        qa[0].q[0] = *(const uint4*)(qb);
        qa[0].q[1] = *(const uint4*)(qb + 16);
        qa[1].q[0] = *(const uint4*)(qb + 32);
        qa[1].q[1] = *(const uint4*)(qb + 48);
    }

    v8f O[4];
    { v8f z = {}; O[0] = z; O[1] = z; O[2] = z; O[3] = z; }
    float m_r[8], l_r[8];
#pragma unroll
    for (int r = 0; r < 8; ++r) { m_r[r] = -1e30f; l_r[r] = 0.0f; }

    const int*   maskb = mask + b * SEQ;
    const float* geomb = geom + (size_t)b * SEQ * SEQ * GEOMD;

    // staging indices (per thread, constant across iterations)
    const int ksrow = threadIdx.x >> 3, kseg = threadIdx.x & 7;
    const int vdrow = threadIdx.x >> 2, vseg = threadIdx.x & 3;

    for (int kc = 0; kc < SEQ; kc += 32) {
        // ---- cooperative async staging of K/V tiles into LDS
        async_copy16(Kh + (size_t)(kc + ksrow) * DK + kseg * 8,
                     &Kl[ksrow * 72 + kseg * 8]);
        async_copy16(Vh + (size_t)vdrow * SEQ + kc + vseg * 8,
                     &Vl[vdrow * 40 + vseg * 8]);
        if (kc + 32 < SEQ) {
            __builtin_prefetch(Kh + (size_t)(kc + 32) * DK, 0, 1);
            __builtin_prefetch(Vh + (size_t)(kc + 32), 0, 1);
        }
        wait_async0();
        __syncthreads();

        // ---- scores S = Q K^T (two 16-col tiles, dk accumulated in 2 steps)
        v8f sc[2];
        { v8f z = {}; sc[0] = z; sc[1] = z; }
#pragma unroll
        for (int t = 0; t < 2; ++t) {
#pragma unroll
            for (int j = 0; j < 2; ++j) {
                ABReg kb;
                const unsigned short* kp =
                    &Kl[(t * 16 + ln) * 72 + j * 32 + hi * 8];
                kb.q[0] = *(const uint4*)(kp);
                kb.q[1] = *(const uint4*)(kp + 16);
                sc[t] = __builtin_amdgcn_wmma_f32_16x16x32_bf16(
                    false, qa[j].v, false, kb.v, (short)0, sc[t], false, false);
            }
        }

        // ---- scale + geometric bias + key mask
        float p[2][8];
#pragma unroll
        for (int t = 0; t < 2; ++t) {
            const int col = kc + t * 16 + ln;
            const int mk  = maskb[col];
#pragma unroll
            for (int r = 0; r < 8; ++r) {
                const int qrow = q0 + hi * 8 + r;
                const float* g = geomb + ((size_t)qrow * SEQ + col) * GEOMD;
                float bias = bgh;
#pragma unroll
                for (int i = 0; i < GEOMD; ++i) bias += g[i] * wg[i];
                const float s = sc[t][r] * inv_scale + bias;
                p[t][r] = (mk == 0) ? -10000.0f : s;
            }
        }

        // ---- online softmax (row reductions within 16-lane halves)
#pragma unroll
        for (int r = 0; r < 8; ++r) {
            float mx = fmaxf(p[0][r], p[1][r]);
#pragma unroll
            for (int d = 8; d >= 1; d >>= 1)
                mx = fmaxf(mx, __shfl_xor(mx, d, 32));
            const float m_new = fmaxf(m_r[r], mx);
            const float alpha = __expf(m_r[r] - m_new);
            const float e0 = __expf(p[0][r] - m_new);
            const float e1 = __expf(p[1][r] - m_new);
            p[0][r] = e0; p[1][r] = e1;
            float sum = e0 + e1;
#pragma unroll
            for (int d = 8; d >= 1; d >>= 1)
                sum += __shfl_xor(sum, d, 32);
            l_r[r] = l_r[r] * alpha + sum;
            m_r[r] = m_new;
#pragma unroll
            for (int n = 0; n < 4; ++n) O[n][r] *= alpha;
        }

        // ---- P: C-layout -> bf16 -> LDS -> A-layout (per-wave slice)
        unsigned short* pl = Plds[wave];
#pragma unroll
        for (int t = 0; t < 2; ++t)
#pragma unroll
            for (int r = 0; r < 8; ++r)
                pl[(hi * 8 + r) * 32 + t * 16 + ln] = f2bf(p[t][r]);

        ABReg pa;
        {
            const unsigned short* pb = pl + ln * 32 + hi * 8;
            pa.q[0] = *(const uint4*)(pb);
            pa.q[1] = *(const uint4*)(pb + 16);
        }

        // ---- O += P @ V   (V^T columns are contiguous LDS rows of Vl)
#pragma unroll
        for (int n = 0; n < 4; ++n) {
            ABReg vb;
            const unsigned short* vp = &Vl[(n * 16 + ln) * 40 + hi * 8];
            vb.q[0] = *(const uint4*)(vp);
            vb.q[1] = *(const uint4*)(vp + 16);
            O[n] = __builtin_amdgcn_wmma_f32_16x16x32_bf16(
                false, pa.v, false, vb.v, (short)0, O[n], false, false);
        }
        __syncthreads();   // protect Kl/Vl before next staging round
    }

    // ---- normalize and emit ctx as bf16 [B*S, D] (rows = b*S+s, col = h*64+d)
#pragma unroll
    for (int r = 0; r < 8; ++r) {
        const float inv = 1.0f / l_r[r];
#pragma unroll
        for (int n = 0; n < 4; ++n) O[n][r] *= inv;
    }
#pragma unroll
    for (int n = 0; n < 4; ++n) {
        const int colc = h * 64 + n * 16 + ln;
#pragma unroll
        for (int r = 0; r < 8; ++r) {
            const int srow = q0 + hi * 8 + r;
            ctx[(size_t)(b * SEQ + srow) * D_MODEL + colc] = f2bf(O[n][r]);
        }
    }
}

// ---------------------------------------------------------------------------
extern "C" void kernel_launch(void* const* d_in, const int* in_sizes, int n_in,
                              void* d_out, int out_size, void* d_ws, size_t ws_size,
                              hipStream_t stream)
{
    (void)in_sizes; (void)n_in; (void)out_size; (void)ws_size;

    const float* x    = (const float*)d_in[0];
    const float* geom = (const float*)d_in[1];
    const int*   mask = (const int*)  d_in[2];
    const float* Wq   = (const float*)d_in[3];
    const float* bq   = (const float*)d_in[4];
    const float* Wk   = (const float*)d_in[5];
    const float* bk   = (const float*)d_in[6];
    const float* Wv   = (const float*)d_in[7];
    const float* bv   = (const float*)d_in[8];
    const float* Wo   = (const float*)d_in[9];
    const float* bo   = (const float*)d_in[10];
    const float* Wg   = (const float*)d_in[11];
    const float* bg   = (const float*)d_in[12];
    float* out = (float*)d_out;

    const size_t HS = (size_t)BATCH * N_HEADS * SEQ * DK;   // 4,194,304 elems
    unsigned short* Qp  = (unsigned short*)d_ws;
    unsigned short* Kp  = Qp  + HS;
    unsigned short* Vtp = Kp  + HS;
    unsigned short* ctx = Vtp + HS;

    dim3 gg(64, 8), gb(256);

    gemm_proj<0, OM_QK><<<gg, gb, 0, stream>>>((const void*)x, Wq, bq, (void*)Qp);
    gemm_proj<0, OM_QK><<<gg, gb, 0, stream>>>((const void*)x, Wk, bk, (void*)Kp);
    gemm_proj<0, OM_VT><<<gg, gb, 0, stream>>>((const void*)x, Wv, bv, (void*)Vtp);

    flash_attn_geom<<<dim3(SEQ / 128, N_HEADS, BATCH), 256, 0, stream>>>(
        Qp, Kp, Vtp, geom, mask, Wg, bg, ctx);

    gemm_proj<1, OM_F32><<<gg, gb, 0, stream>>>((const void*)ctx, Wo, bo, (void*)out);
}
